// Atten2Map_50835232916100
// MI455X (gfx1250) — compile-verified, matrix operand order checked by reading, not run on hardware
//
#include <hip/hip_runtime.h>

#define NNEI 128
#define NI   128
#define NDIM 32
#define NH   8
#define NCOL 512   // NDIM * 2 * NH

typedef __attribute__((ext_vector_type(16))) __bf16 v16bf;
typedef __attribute__((ext_vector_type(8)))  float  v8f;
typedef __attribute__((ext_vector_type(4)))  float  f32x4;

union FragBF { v16bf v; uint4 q[2]; };

__device__ __forceinline__ unsigned short f2bf(float f) {
  unsigned int u = __float_as_uint(f);
  u += 0x7FFFu + ((u >> 16) & 1u);
  return (unsigned short)(u >> 16);
}

// Load a 16-bit WMMA operand fragment: two contiguous 16B chunks at p and p+16
// (per-lane K-halves 0..7 / 16..23 for lanes 0-15, 8..15 / 24..31 for lanes
//  16-31; caller adds lane_half*8 into p).
__device__ __forceinline__ v16bf ld_frag16(const unsigned short* p) {
  FragBF f;
  f.q[0] = *(const uint4*)(p);
  f.q[1] = *(const uint4*)(p + 16);
  return f.v;
}

// ---- kernel 1: W (128x512 f32, k-major) -> Wt (512x128 bf16, n-major) ----
__global__ void w_transpose_bf16(const float* __restrict__ W,
                                 unsigned short* __restrict__ Wt) {
  int idx = blockIdx.x * blockDim.x + threadIdx.x;   // 0..65535
  int k = idx >> 9;
  int n = idx & 511;
  Wt[n * NI + k] = f2bf(W[idx]);
}

// LDS row strides (elements), padded so row-parallel accesses rotate banks by 4
#define SA_STRIDE 136     // 272 B rows
#define QK_STRIDE 520     // 1040 B rows
#define SA_ELEMS  (NNEI * SA_STRIDE)
#define QK_ELEMS  (NNEI * QK_STRIDE)

__global__ void __launch_bounds__(256, 1)
atten2map_kernel(const float* __restrict__ g2,
                 const float* __restrict__ h2g,
                 const unsigned char* __restrict__ nmask,
                 const float* __restrict__ swg,
                 const unsigned short* __restrict__ Wt,
                 float* __restrict__ out) {
  extern __shared__ char smem_raw[];
  unsigned short* sA  = (unsigned short*)smem_raw;                      // g2 tile bf16
  unsigned short* sQK = (unsigned short*)(smem_raw + SA_ELEMS * 2);     // g2qk bf16
  float* s_sw   = (float*)(smem_raw + SA_ELEMS * 2 + QK_ELEMS * 2);
  float* s_mask = s_sw + NNEI;
  float* s_h2   = s_mask + NNEI;   // [128][3]

  const int tid  = threadIdx.x;
  const int lane = tid & 31;
  const int wave = tid >> 5;
  const int ln   = lane & 15;
  const int lh   = lane >> 4;

  const size_t blk = blockIdx.x;                       // b*nloc + l
  const f32x4* g2p = (const f32x4*)(g2 + blk * (size_t)(NNEI * NI));

  // ---- stage 0: stage g2 tile (f32 -> bf16, nontemporal read) + metadata ----
  #pragma unroll
  for (int i = 0; i < 16; ++i) {
    int lin = tid + i * 256;           // float4 index 0..4095
    int row = lin >> 5;                // 32 float4 per row
    int c4  = lin & 31;
    f32x4 v = __builtin_nontemporal_load(&g2p[lin]);
    unsigned int lo = (unsigned int)f2bf(v.x) | ((unsigned int)f2bf(v.y) << 16);
    unsigned int hi = (unsigned int)f2bf(v.z) | ((unsigned int)f2bf(v.w) << 16);
    *(uint2*)&sA[row * SA_STRIDE + c4 * 4] = make_uint2(lo, hi);
  }
  if (tid < NNEI) {
    s_sw[tid]   = swg[blk * NNEI + tid];
    s_mask[tid] = nmask[blk * NNEI + tid] ? 1.0f : 0.0f;
    s_h2[tid * 3 + 0] = h2g[blk * (size_t)(NNEI * 3) + tid * 3 + 0];
    s_h2[tid * 3 + 1] = h2g[blk * (size_t)(NNEI * 3) + tid * 3 + 1];
    s_h2[tid * 3 + 2] = h2g[blk * (size_t)(NNEI * 3) + tid * 3 + 2];
  }
  __syncthreads();

  // ---- stage 1: g2qk = g2 @ W  (128x128x512, bf16 WMMA, f32 acc) ----
  // wave owns 4 of 32 N-tiles; B fragments from global Wt (L2-hot)
  for (int ntl = 0; ntl < 4; ++ntl) {
    int nt  = wave * 4 + ntl;
    int col = nt * 16 + ln;                       // output col == Wt row
    const unsigned short* wrow = Wt + (size_t)col * NI + lh * 8;
    v16bf B[4];
    #pragma unroll
    for (int kk = 0; kk < 4; ++kk) B[kk] = ld_frag16(wrow + kk * 32);
    for (int mt = 0; mt < 8; ++mt) {
      const unsigned short* arow = sA + (mt * 16 + ln) * SA_STRIDE + lh * 8;
      v8f acc = {};
      #pragma unroll
      for (int kk = 0; kk < 4; ++kk) {
        v16bf A = ld_frag16(arow + kk * 32);
        acc = __builtin_amdgcn_wmma_f32_16x16x32_bf16(
                  false, A, false, B[kk], (short)0, acc, false, false);
      }
      #pragma unroll
      for (int r = 0; r < 8; ++r) {
        int row = mt * 16 + r + 8 * lh;           // D layout: vgpr r -> M r / r+8
        sQK[row * QK_STRIDE + nt * 16 + ln] = f2bf(acc[r]);
      }
    }
  }
  __syncthreads();

  // ---- stage 2: per-head attention (wave == head) ----
  const int h = wave;
  const unsigned short* qbase = sQK + h * NDIM;             // q cols [h*32 .. )
  const unsigned short* kbase = sQK + NH * NDIM + h * NDIM; // k cols [256+h*32 .. )
  const float inv_sqrt_d = 0.17677669529663687f;            // 1/sqrt(32)
  const float inv_sqrt_3 = 0.57735026918962576f;

  // K^T fragments are M-tile invariant: load the whole 32x128 operand once.
  v16bf K[8];
  #pragma unroll
  for (int nt = 0; nt < 8; ++nt)
    K[nt] = ld_frag16(kbase + (nt * 16 + ln) * QK_STRIDE + lh * 8);

  float swn[8], mkn[8], hn0[8], hn1[8], hn2[8];             // column side (n = nt*16+ln)
  #pragma unroll
  for (int nt = 0; nt < 8; ++nt) {
    int n = nt * 16 + ln;
    swn[nt] = s_sw[n];
    mkn[nt] = s_mask[n];
    hn0[nt] = s_h2[n * 3 + 0];
    hn1[nt] = s_h2[n * 3 + 1];
    hn2[nt] = s_h2[n * 3 + 2];
  }

  float* outp = out + blk * (size_t)(NNEI * NNEI * NH) + h;

  for (int mt = 0; mt < 8; ++mt) {
    v16bf Q = ld_frag16(qbase + (mt * 16 + ln) * QK_STRIDE + lh * 8);
    v8f S[8];
    #pragma unroll
    for (int nt = 0; nt < 8; ++nt) {
      v8f z = {};
      S[nt] = __builtin_amdgcn_wmma_f32_16x16x32_bf16(
                  false, Q, false, K[nt], (short)0, z, false, false);
    }
    #pragma unroll
    for (int r = 0; r < 8; ++r) {
      int m = mt * 16 + r + 8 * lh;
      float x[8];
      float xmax = -3.0e38f;
      #pragma unroll
      for (int nt = 0; nt < 8; ++nt) {
        float v = (mkn[nt] != 0.0f) ? S[nt][r] * inv_sqrt_d : -3.0e38f;
        x[nt] = v;
        xmax = fmaxf(xmax, v);
      }
      // row reduction: each 16-lane half holds one M row across all n
      #pragma unroll
      for (int d = 1; d < 16; d <<= 1) xmax = fmaxf(xmax, __shfl_xor(xmax, d, 32));
      float ssum = 0.0f;
      #pragma unroll
      for (int nt = 0; nt < 8; ++nt) {
        float e = (mkn[nt] != 0.0f) ? __expf(x[nt] - xmax) : 0.0f;
        x[nt] = e;
        ssum += e;
      }
      #pragma unroll
      for (int d = 1; d < 16; d <<= 1) ssum += __shfl_xor(ssum, d, 32);
      float rowf = s_mask[m] * s_sw[m] * __builtin_amdgcn_rcpf(ssum);
      float hm0 = s_h2[m * 3 + 0], hm1 = s_h2[m * 3 + 1], hm2 = s_h2[m * 3 + 2];
      float* orow = outp + (size_t)m * (NNEI * NH);
      #pragma unroll
      for (int nt = 0; nt < 8; ++nt) {
        int n = nt * 16 + ln;
        float dot = (hm0 * hn0[nt] + hm1 * hn1[nt] + hm2 * hn2[nt]) * inv_sqrt_3;
        orow[(size_t)n * NH] = x[nt] * rowf * swn[nt] * dot;
      }
    }
  }
}

extern "C" void kernel_launch(void* const* d_in, const int* in_sizes, int n_in,
                              void* d_out, int out_size, void* d_ws, size_t ws_size,
                              hipStream_t stream) {
  const float* g2          = (const float*)d_in[0];
  const float* h2          = (const float*)d_in[1];
  const unsigned char* msk = (const unsigned char*)d_in[2];
  const float* sw          = (const float*)d_in[3];
  const float* W           = (const float*)d_in[4];
  unsigned short* Wt = (unsigned short*)d_ws;     // 512*128 bf16 = 128 KB

  w_transpose_bf16<<<(NI * NCOL) / 256, 256, 0, stream>>>(W, Wt);

  size_t smem = (size_t)SA_ELEMS * 2 + (size_t)QK_ELEMS * 2
              + (size_t)(NNEI + NNEI + NNEI * 3) * sizeof(float);
  atten2map_kernel<<<4 * 256, 256, smem, stream>>>(g2, h2, msk, sw, Wt, (float*)d_out);
}
